// CosmoMamba_65420941853163
// MI455X (gfx1250) — compile-verified, matrix operand order checked by reading, not run on hardware
//
#include <hip/hip_runtime.h>
#include <hip/hip_bf16.h>
#include <cstdint>

// ---------------------------------------------------------------------------
// CosmoMamba forward for MI455X (gfx1250): bf16 WMMA GEMMs (L2-resident
// weights, 173MB < 192MB L2), float4/b128 data movement, channel-parallel
// selective scan with async global->LDS staging of the shared proj rows.
// ---------------------------------------------------------------------------

#define B_SZ   8
#define IMG    256
#define EMB    384
#define DI     768      // EMB*2
#define DS     16
#define DCONV  4
#define FFN    1536     // EMB*4
#define LSEQ   256      // 16x16 tokens
#define NTOK   (B_SZ*LSEQ)      // 2048
#define BN1C   192
#define CH1HW  64
#define BN_SCALE 0.9999950000374997f   // 1/sqrt(1+1e-5)

typedef __attribute__((ext_vector_type(16))) __bf16 v16bf;
typedef __attribute__((ext_vector_type(4)))  __bf16 v4bf;
typedef __attribute__((ext_vector_type(4)))  float  v4f;
typedef __attribute__((ext_vector_type(8)))  float  v8f;

// ---- helpers ----------------------------------------------------------------
__device__ __forceinline__ float gelu_exact(float x) {
  return 0.5f * x * (1.0f + erff(x * 0.70710678118654752f));
}
__device__ __forceinline__ float silu(float x) {
  return x / (1.0f + __expf(-x));
}
__device__ __forceinline__ float softplus(float x) {
  return (x > 20.f) ? x : log1pf(__expf(x));
}
__device__ __forceinline__ int perm2(int t) {        // 16x16 transpose (involution)
  return ((t & 15) << 4) | (t >> 4);
}
// WMMA 16-bit A/B fragment K index for element i, lane-half hi (ISA 7.12.2)
__device__ __forceinline__ int kmap(int i, int hi) {
  return (i < 8) ? (hi * 8 + i) : (16 + hi * 8 + (i - 8));
}

// ---- async global->LDS (CDNA5), guarded so compilation never breaks ---------
// Probe result: builtin exists; param 0 is a non-const AS(1) int*.
__device__ __forceinline__ void async_ld32(const float* g, float* l) {
#if __has_builtin(__builtin_amdgcn_global_load_async_to_lds_b32)
  float* gnc = const_cast<float*>(g);
  __builtin_amdgcn_global_load_async_to_lds_b32(
      (__attribute__((address_space(1))) int*)gnc,
      (__attribute__((address_space(3))) int*)l, 0, 0);
#else
  *l = *g;
#endif
}
__device__ __forceinline__ void async_wait0() {
#if __has_builtin(__builtin_amdgcn_global_load_async_to_lds_b32)
#if __has_builtin(__builtin_amdgcn_s_wait_asynccnt)
  __builtin_amdgcn_s_wait_asynccnt(0);
#else
  asm volatile("s_wait_asynccnt 0x0" ::: "memory");
#endif
#endif
}

// ---- tiled bf16 WMMA GEMM:  C[M,N] (+)= act(A[M,K] @ W[N,K]^T + bias) -------
// block = 256 thr = 8 waves; block tile 128x64; wave tile 32x32 (2x2 wmma)
#define BM 128
#define BN 64
#define BK 32
#define LDPAD 4   // row stride 36 bf16 = 72B, keeps 8B stores aligned

template<int ACT, bool BIAS, bool RES>
__global__ __launch_bounds__(256)
void k_gemm(const float* __restrict__ A, const float* __restrict__ W,
            const float* __restrict__ bias, float* __restrict__ C,
            int M, int N, int K, int ldc)
{
  __shared__ __bf16 As[BM][BK + LDPAD];
  __shared__ __bf16 Bs[BN][BK + LDPAD];
  const int tid  = threadIdx.x;
  const int wave = tid >> 5;
  const int lane = tid & 31;
  const int wm   = wave & 3;        // 4 M-bands of 32
  const int wn   = wave >> 2;       // 2 N-bands of 32
  const int m0   = blockIdx.y * BM;
  const int n0   = blockIdx.x * BN;
  const int lm   = lane & 15;
  const int hi   = lane >> 4;

  v8f acc[2][2] = {};

  for (int k0 = 0; k0 < K; k0 += BK) {
    // stage A tile (128x32): 4 x float4 per thread, fptrunc -> bf16x4 store
    #pragma unroll
    for (int i = 0; i < 4; ++i) {
      int e = i * 256 + tid;              // 1024 float4s
      int r = e >> 3, c4 = (e & 7) * 4;
      const v4f v = *(const v4f*)&A[(size_t)(m0 + r) * K + (k0 + c4)];
      *(v4bf*)&As[r][c4] = __builtin_convertvector(v, v4bf);
    }
    // stage B tile (64x32): 2 x float4 per thread
    #pragma unroll
    for (int i = 0; i < 2; ++i) {
      int e = i * 256 + tid;              // 512 float4s
      int r = e >> 3, c4 = (e & 7) * 4;
      const v4f v = *(const v4f*)&W[(size_t)(n0 + r) * K + (k0 + c4)];
      *(v4bf*)&Bs[r][c4] = __builtin_convertvector(v, v4bf);
    }
    // prefetch next K tile of A into the cache hierarchy
    if (k0 + BK < K)
      __builtin_prefetch(&A[(size_t)(m0 + (tid >> 1)) * K + (k0 + BK)], 0, 1);
    __syncthreads();

    v16bf af[2], bf[2];
    #pragma unroll
    for (int f = 0; f < 2; ++f) {
      int mr = wm * 32 + f * 16 + lm;
      int nr = wn * 32 + f * 16 + lm;
      #pragma unroll
      for (int i = 0; i < 16; ++i) {
        int kk = kmap(i, hi);
        af[f][i] = As[mr][kk];
        bf[f][i] = Bs[nr][kk];
      }
    }
    #pragma unroll
    for (int mf = 0; mf < 2; ++mf)
      #pragma unroll
      for (int nf = 0; nf < 2; ++nf)
        acc[mf][nf] = __builtin_amdgcn_wmma_f32_16x16x32_bf16(
            false, af[mf], false, bf[nf], (short)0, acc[mf][nf], false, false);
    __syncthreads();
  }

  // epilogue (C/D layout: vgpr r -> M = r + 8*hi)
  #pragma unroll
  for (int mf = 0; mf < 2; ++mf)
    #pragma unroll
    for (int nf = 0; nf < 2; ++nf) {
      int n  = n0 + wn * 32 + nf * 16 + lm;
      float bv = BIAS ? bias[n] : 0.f;
      #pragma unroll
      for (int r = 0; r < 8; ++r) {
        int m = m0 + wm * 32 + mf * 16 + r + 8 * hi;
        float v = acc[mf][nf][r] + bv;
        if (ACT == 1) v = gelu_exact(v);
        float* cp = &C[(size_t)m * ldc + n];
        if (RES) v += *cp;
        *cp = v;
      }
    }
}

// ---- patch embedding --------------------------------------------------------
__global__ __launch_bounds__(256)
void k_conv1(const float* __restrict__ x, const float* __restrict__ w,
             const float* __restrict__ g, const float* __restrict__ bb,
             float* __restrict__ out)
{
  int idx = blockIdx.x * 256 + threadIdx.x;             // 8*192*64*64
  int ow = idx & 63, oh = (idx >> 6) & 63;
  int c  = (idx >> 12) % BN1C;
  int b  = idx / (4096 * BN1C);
  float acc = 0.f;
  const float* wr = w + c * 49;
  const float* xb = x + (size_t)b * IMG * IMG;
  #pragma unroll
  for (int kh = 0; kh < 7; ++kh) {
    int ih = oh * 4 - 3 + kh;
    if ((unsigned)ih >= IMG) continue;
    #pragma unroll
    for (int kw = 0; kw < 7; ++kw) {
      int iw = ow * 4 - 3 + kw;
      if ((unsigned)iw >= IMG) continue;
      acc += wr[kh * 7 + kw] * xb[ih * IMG + iw];
    }
  }
  out[idx] = gelu_exact(BN_SCALE * acc * g[c] + bb[c]);
}

__global__ __launch_bounds__(256)
void k_conv2(const float* __restrict__ c1, const float* __restrict__ w,
             const float* __restrict__ g, const float* __restrict__ bb,
             const float* __restrict__ pos, float* __restrict__ t)
{
  int idx = blockIdx.x * 256 + threadIdx.x;             // 8*256*384
  int oc = idx % EMB;
  int tk = (idx / EMB) % LSEQ;
  int b  = idx / (EMB * LSEQ);
  int ow = tk & 15, oh = tk >> 4;
  float acc = 0.f;
  const float* wr = w + (size_t)oc * BN1C * 9;
  for (int ic = 0; ic < BN1C; ++ic) {
    const float* xp = c1 + ((size_t)b * BN1C + ic) * CH1HW * CH1HW;
    #pragma unroll
    for (int kh = 0; kh < 3; ++kh) {
      int ih = oh * 4 - 1 + kh;
      if ((unsigned)ih >= CH1HW) continue;
      #pragma unroll
      for (int kw = 0; kw < 3; ++kw) {
        int iw = ow * 4 - 1 + kw;
        if ((unsigned)iw >= CH1HW) continue;
        acc += wr[ic * 9 + kh * 3 + kw] * xp[ih * CH1HW + iw];
      }
    }
  }
  t[idx] = gelu_exact(BN_SCALE * acc * g[oc] + bb[oc]) + pos[tk * EMB + oc];
}

// ---- LayerNorm (one wave32 per row, float4 lanes), optional SiLU(z) gate ----
__global__ __launch_bounds__(256)
void k_ln(const float* __restrict__ in, float* __restrict__ out,
          const float* __restrict__ g, const float* __restrict__ b,
          const float* __restrict__ zg, int zld, int D)
{
  int row  = blockIdx.x * 8 + (threadIdx.x >> 5);
  int lane = threadIdx.x & 31;
  const float* x = in + (size_t)row * D;
  float s = 0.f, s2 = 0.f;
  for (int i = lane * 4; i < D; i += 128) {
    float4 v = *(const float4*)&x[i];
    s  += v.x + v.y + v.z + v.w;
    s2 += v.x*v.x + v.y*v.y + v.z*v.z + v.w*v.w;
  }
  #pragma unroll
  for (int off = 16; off; off >>= 1) {
    s  += __shfl_xor(s,  off, 32);
    s2 += __shfl_xor(s2, off, 32);
  }
  float m  = s / D;
  float iv = rsqrtf(s2 / D - m * m + 1e-5f);
  float* o = out + (size_t)row * D;
  const float* z = zg ? zg + (size_t)row * zld : nullptr;
  for (int i = lane * 4; i < D; i += 128) {
    float4 v  = *(const float4*)&x[i];
    float4 gv = *(const float4*)&g[i];
    float4 bv = *(const float4*)&b[i];
    float4 r;
    r.x = (v.x - m) * iv * gv.x + bv.x;
    r.y = (v.y - m) * iv * gv.y + bv.y;
    r.z = (v.z - m) * iv * gv.z + bv.z;
    r.w = (v.w - m) * iv * gv.w + bv.w;
    if (z) {
      float4 zv = *(const float4*)&z[i];
      r.x *= silu(zv.x); r.y *= silu(zv.y); r.z *= silu(zv.z); r.w *= silu(zv.w);
    }
    *(float4*)&o[i] = r;
  }
}

// ---- direction gather / scatter over 16x16 token grid (float4) --------------
__global__ __launch_bounds__(256)
void k_gather(const float* __restrict__ xin, float* __restrict__ xout, int dir)
{
  int idx = blockIdx.x * 256 + threadIdx.x;             // 2048*96
  int c4 = (idx % 96) * 4, row = idx / 96;
  int b = row >> 8, p = row & 255;
  int l;
  if      (dir == 0) l = p;
  else if (dir == 1) l = 255 - p;
  else if (dir == 2) l = perm2(p);
  else               l = perm2(255 - p);
  *(float4*)&xout[(size_t)row * EMB + c4] =
      *(const float4*)&xin[(size_t)((b << 8) + l) * EMB + c4];
}

__global__ __launch_bounds__(256)
void k_scatter(const float* __restrict__ yp, float* __restrict__ ycat, int dir)
{
  int idx = blockIdx.x * 256 + threadIdx.x;             // 2048*96
  int c4 = (idx % 96) * 4, row = idx / 96;
  int b = row >> 8, t = row & 255;
  int gsrc;
  if      (dir == 0) gsrc = t;
  else if (dir == 1) gsrc = 255 - t;
  else if (dir == 2) gsrc = perm2(t);
  else               gsrc = 255 - perm2(t);
  *(float4*)&ycat[(size_t)row * (4 * EMB) + dir * EMB + c4] =
      *(const float4*)&yp[(size_t)((b << 8) + gsrc) * EMB + c4];
}

// ---- causal depthwise conv(4) + SiLU on xz[:, :DI] --------------------------
__global__ __launch_bounds__(256)
void k_dwconv(const float* __restrict__ xz, const float* __restrict__ w,
              const float* __restrict__ cb, float* __restrict__ xb)
{
  int idx = blockIdx.x * 256 + threadIdx.x;             // 2048*768
  int d = idx % DI, row = idx / DI;
  int b = row >> 8, p = row & 255;
  float4 wv = *(const float4*)&w[d * DCONV];
  const float wj[4] = { wv.x, wv.y, wv.z, wv.w };
  float acc = cb[d];
  #pragma unroll
  for (int j = 0; j < DCONV; ++j) {
    int q = p - (DCONV - 1) + j;
    if (q >= 0)
      acc += wj[j] * xz[(size_t)((b << 8) + q) * (2 * DI) + d];
  }
  xb[idx] = silu(acc);
}

// ---- x_proj: proj[2048,33] = xb @ x_proj^T (one wave per output scalar) -----
__global__ __launch_bounds__(256)
void k_xproj(const float* __restrict__ xb, const float* __restrict__ w,
             float* __restrict__ proj)
{
  int gw = blockIdx.x * 8 + (threadIdx.x >> 5);
  if (gw >= NTOK * 33) return;
  int row = gw / 33, o = gw % 33;
  int lane = threadIdx.x & 31;
  const float* a  = xb + (size_t)row * DI;
  const float* ww = w + (size_t)o * DI;
  float s = 0.f;
  #pragma unroll
  for (int k = lane * 4; k < DI; k += 128) {
    float4 av = *(const float4*)&a[k];
    float4 wv = *(const float4*)&ww[k];
    s += av.x*wv.x + av.y*wv.y + av.z*wv.z + av.w*wv.w;
  }
  #pragma unroll
  for (int off = 16; off; off >>= 1) s += __shfl_xor(s, off, 32);
  if (lane == 0) proj[(size_t)row * 33 + o] = s;
}

// ---- selective scan: one thread per (batch, channel); h[16] in registers ----
// Each 256-thread block covers one batch (DI = 3*256), so all threads share
// the per-timestep proj row -> stage it in LDS with double-buffered async
// global->LDS copies (ASYNCcnt).
__global__ __launch_bounds__(256)
void k_scan(const float* __restrict__ proj, const float* __restrict__ xb,
            const float* __restrict__ dtw, const float* __restrict__ dtb,
            const float* __restrict__ Alog, const float* __restrict__ Dp,
            float* __restrict__ y)
{
  __shared__ float prbuf[2][40];
  const int tid  = threadIdx.x;
  const int gtid = blockIdx.x * 256 + tid;              // 24 blocks x 256 = 6144
  const int b = gtid / DI, d = gtid % DI;

  float A[DS];
  #pragma unroll
  for (int s = 0; s < DS; ++s) A[s] = -__expf(Alog[d * DS + s]);
  const float dw = dtw[d], db = dtb[d], Dv = Dp[d];

  if (tid < 33) async_ld32(&proj[(size_t)(b << 8) * 33 + tid], &prbuf[0][tid]);
  async_wait0();
  __syncthreads();

  float h[DS] = {};
  for (int l = 0; l < LSEQ; ++l) {
    if (l + 1 < LSEQ && tid < 33)
      async_ld32(&proj[(size_t)((b << 8) + l + 1) * 33 + tid],
                 &prbuf[(l + 1) & 1][tid]);
    const float* pr = prbuf[l & 1];
    const int row = (b << 8) + l;
    float dt = softplus(pr[0] * dw + db);
    float xv = xb[(size_t)row * DI + d];
    float xd = xv * dt;
    float acc = 0.f;
    #pragma unroll
    for (int s = 0; s < DS; ++s) {
      float dA = __expf(dt * A[s]);
      h[s] = dA * h[s] + xd * pr[1 + s];
      acc += h[s] * pr[17 + s];
    }
    y[(size_t)row * DI + d] = acc + xv * Dv;
    async_wait0();
    __syncthreads();
  }
}

// ---- final pool / head / outputs --------------------------------------------
__global__ __launch_bounds__(256)
void k_pool(const float* __restrict__ tn, float* __restrict__ pooled)
{
  int idx = blockIdx.x * 256 + threadIdx.x;             // 8*96
  if (idx >= B_SZ * (EMB / 4)) return;
  int b = idx / (EMB / 4), c4 = (idx % (EMB / 4)) * 4;
  float4 s = {0.f, 0.f, 0.f, 0.f};
  for (int t = 0; t < LSEQ; ++t) {
    float4 v = *(const float4*)&tn[(size_t)((b << 8) + t) * EMB + c4];
    s.x += v.x; s.y += v.y; s.z += v.z; s.w += v.w;
  }
  const float inv = 1.0f / LSEQ;
  s.x *= inv; s.y *= inv; s.z *= inv; s.w *= inv;
  *(float4*)&pooled[b * EMB + c4] = s;
}

__global__ __launch_bounds__(256)
void k_head(const float* __restrict__ pooled, const float* __restrict__ w,
            const float* __restrict__ bb, float* __restrict__ h)
{
  int idx = blockIdx.x * 256 + threadIdx.x;             // 8*384
  int b = idx / EMB, c = idx % EMB;
  float acc = bb[c];
  for (int k = 0; k < EMB; k += 4) {
    float4 pv = *(const float4*)&pooled[b * EMB + k];
    float4 wv = *(const float4*)&w[c * EMB + k];
    acc += pv.x*wv.x + pv.y*wv.y + pv.z*wv.z + pv.w*wv.w;
  }
  h[idx] = gelu_exact(acc);
}

__global__ void k_out(const float* __restrict__ h,
                      const float* __restrict__ mw, const float* __restrict__ mb,
                      const float* __restrict__ lw, const float* __restrict__ lb,
                      float* __restrict__ out)
{
  int tid = threadIdx.x;                                // 32 threads
  int sel = tid >> 4, r = tid & 15;
  int b = r >> 1, j = r & 1;
  const float* w  = sel ? lw : mw;
  const float* bi = sel ? lb : mb;
  float acc = bi[j];
  for (int k = 0; k < EMB; ++k)
    acc += h[b * EMB + k] * w[j * EMB + k];
  out[sel * (B_SZ * 2) + b * 2 + j] = acc;
}

// ---------------------------------------------------------------------------
extern "C" void kernel_launch(void* const* d_in, const int* in_sizes, int n_in,
                              void* d_out, int out_size, void* d_ws, size_t ws_size,
                              hipStream_t stream)
{
  (void)in_sizes; (void)n_in; (void)out_size; (void)ws_size;
  auto IN = [&](int i) { return (const float*)d_in[i]; };

  // input layout (setup_inputs dict order, nested dicts flattened in order)
  const float* X        = IN(0);
  const float* conv1_w  = IN(1);
  const float* bn1_g    = IN(2);
  const float* bn1_b    = IN(3);
  const float* conv2_w  = IN(4);
  const float* bn2_g    = IN(5);
  const float* bn2_b    = IN(6);
  const float* posemb   = IN(7);
  const int BLK0 = 8, BLKSZ = 54, SSMSZ = 11;
  const int POST = BLK0 + 8 * BLKSZ;                    // 440
  const float* norm_g = IN(POST + 0);
  const float* norm_b = IN(POST + 1);
  const float* head_w = IN(POST + 2);
  const float* head_b = IN(POST + 3);
  const float* mean_w = IN(POST + 4);
  const float* mean_b = IN(POST + 5);
  const float* lvar_w = IN(POST + 6);
  const float* lvar_b = IN(POST + 7);

  // workspace layout (floats)
  float* ws = (float*)d_ws;
  size_t o = 0;
  float* c1    = ws + o; o += (size_t)B_SZ * BN1C * CH1HW * CH1HW;  // 6.29M
  float* t     = ws + o; o += (size_t)NTOK * EMB;                   // 0.79M
  float* lnb   = ws + o; o += (size_t)NTOK * EMB;
  float* xg    = ws + o; o += (size_t)NTOK * EMB;
  float* xz    = ws + o; o += (size_t)NTOK * 2 * DI;                // 3.15M
  float* xb    = ws + o; o += (size_t)NTOK * DI;
  float* proj  = ws + o; o += (size_t)NTOK * 33;
  float* yd    = ws + o; o += (size_t)NTOK * DI;
  float* yp    = ws + o; o += (size_t)NTOK * EMB;
  float* ycat  = ws + o; o += (size_t)NTOK * 4 * EMB;
  float* hffn  = ws + o; o += (size_t)NTOK * FFN;
  float* pooled= ws + o; o += (size_t)B_SZ * EMB;
  float* hbuf  = ws + o; o += (size_t)B_SZ * EMB;

  // ---- patch embed ----
  k_conv1<<<(B_SZ*BN1C*CH1HW*CH1HW)/256, 256, 0, stream>>>(X, conv1_w, bn1_g, bn1_b, c1);
  k_conv2<<<(NTOK*EMB)/256, 256, 0, stream>>>(c1, conv2_w, bn2_g, bn2_b, posemb, t);

  // ---- blocks ----
  for (int blk = 0; blk < 8; ++blk) {
    int bp = BLK0 + blk * BLKSZ;
    const float* ln1_g = IN(bp + 0);
    const float* ln1_b = IN(bp + 1);

    k_ln<<<NTOK/8, 256, 0, stream>>>(t, lnb, ln1_g, ln1_b, nullptr, 0, EMB);

    for (int dir = 0; dir < 4; ++dir) {
      int sb = bp + 2 + dir * SSMSZ;
      const float* in_proj = IN(sb + 0);
      const float* conv_w  = IN(sb + 1);
      const float* conv_b  = IN(sb + 2);
      const float* x_proj  = IN(sb + 3);
      const float* dt_w    = IN(sb + 4);
      const float* dt_b    = IN(sb + 5);
      const float* A_log   = IN(sb + 6);
      const float* Dp      = IN(sb + 7);
      const float* out_pj  = IN(sb + 8);
      const float* sln_g   = IN(sb + 9);
      const float* sln_b   = IN(sb + 10);

      k_gather<<<(NTOK*(EMB/4))/256, 256, 0, stream>>>(lnb, xg, dir);
      k_gemm<0,false,false><<<dim3((2*DI)/BN, NTOK/BM), 256, 0, stream>>>(
          xg, in_proj, nullptr, xz, NTOK, 2*DI, EMB, 2*DI);
      k_dwconv<<<(NTOK*DI)/256, 256, 0, stream>>>(xz, conv_w, conv_b, xb);
      k_xproj<<<(NTOK*33 + 7)/8, 256, 0, stream>>>(xb, x_proj, proj);
      k_scan<<<(B_SZ*DI)/256, 256, 0, stream>>>(proj, xb, dt_w, dt_b, A_log, Dp, yd);
      k_ln<<<NTOK/8, 256, 0, stream>>>(yd, yd, sln_g, sln_b, xz + DI, 2*DI, DI);
      k_gemm<0,false,false><<<dim3(EMB/BN, NTOK/BM), 256, 0, stream>>>(
          yd, out_pj, nullptr, yp, NTOK, EMB, DI, EMB);
      k_scatter<<<(NTOK*(EMB/4))/256, 256, 0, stream>>>(yp, ycat, dir);
    }

    const float* merge_w = IN(bp + 46);
    const float* merge_b = IN(bp + 47);
    const float* ln2_g   = IN(bp + 48);
    const float* ln2_b   = IN(bp + 49);
    const float* ffn_w1  = IN(bp + 50);
    const float* ffn_b1  = IN(bp + 51);
    const float* ffn_w2  = IN(bp + 52);
    const float* ffn_b2  = IN(bp + 53);

    // t = t + ycat @ merge_w^T + merge_b   (residual)
    k_gemm<0,true,true><<<dim3(EMB/BN, NTOK/BM), 256, 0, stream>>>(
        ycat, merge_w, merge_b, t, NTOK, EMB, 4*EMB, EMB);

    k_ln<<<NTOK/8, 256, 0, stream>>>(t, lnb, ln2_g, ln2_b, nullptr, 0, EMB);
    k_gemm<1,true,false><<<dim3(FFN/BN, NTOK/BM), 256, 0, stream>>>(
        lnb, ffn_w1, ffn_b1, hffn, NTOK, FFN, EMB, FFN);
    k_gemm<0,true,true><<<dim3(EMB/BN, NTOK/BM), 256, 0, stream>>>(
        hffn, ffn_w2, ffn_b2, t, NTOK, EMB, FFN, EMB);
  }

  // ---- final norm / pool / heads ----
  k_ln<<<NTOK/8, 256, 0, stream>>>(t, lnb, norm_g, norm_b, nullptr, 0, EMB);
  k_pool<<<(B_SZ*(EMB/4) + 255)/256, 256, 0, stream>>>(lnb, pooled);
  k_head<<<(B_SZ*EMB)/256, 256, 0, stream>>>(pooled, head_w, head_b, hbuf);
  k_out<<<1, 32, 0, stream>>>(hbuf, mean_w, mean_b, lvar_w, lvar_b, (float*)d_out);
}
